// GAT_CORA_14740327760225
// MI455X (gfx1250) — matrix operand order repaired and checked by reading, not compiled, not run on hardware
//
#include <hip/hip_runtime.h>
#include <hip/hip_bf16.h>
#include <math.h>

typedef __bf16 bhalf;
typedef __attribute__((ext_vector_type(16))) __bf16 v16bf;
typedef __attribute__((ext_vector_type(8)))  __bf16 v8bf;
typedef __attribute__((ext_vector_type(8)))  float  v8f;
// 4-float vector with only 4-byte alignment (x rows are 1433 floats -> rows
// are 4B-aligned only); lets the backend emit unaligned b128 loads.
typedef float f4u __attribute__((ext_vector_type(4), aligned(4)));

#define F_IN   1433
#define HID    8
#define H1     8
#define NCLS   7
#define NEG_SLOPE 0.2f

// ---------------------------------------------------------------------------
// Utility: float atomic max via sign-aware integer atomics (monotone mapping)
// ---------------------------------------------------------------------------
__device__ __forceinline__ void atomicMaxF32(float* addr, float val) {
    if (val >= 0.0f) {
        atomicMax((int*)addr, __float_as_int(val));
    } else {
        atomicMin((unsigned int*)addr, __float_as_uint(val));
    }
}

__device__ __forceinline__ float leaky(float v) {
    return v > 0.0f ? v : NEG_SLOPE * v;
}

// ---------------------------------------------------------------------------
// Fill kernel (init accumulators / running-max buffers)
// ---------------------------------------------------------------------------
__global__ void fill_kernel(float* __restrict__ p, long long n, float v) {
    long long i = (long long)blockIdx.x * blockDim.x + threadIdx.x;
    if (i < n) p[i] = v;
}

// ---------------------------------------------------------------------------
// Layer-1 projection: h1 = x @ W1   (N x 1433) @ (1433 x 64) -> fp32
// bf16 WMMA with fp32 accumulation. 128 rows per block (8 waves x 16 rows),
// each wave computes a 16x64 tile via 4 accumulators of v8f.
// K loop split: 44 unguarded full tiles + 1 guarded tail tile.
// ---------------------------------------------------------------------------
#define TILE_M 128
#define KSTEP  32

__global__ __launch_bounds__(256) void gemm1_wmma(
        const float* __restrict__ x, const float* __restrict__ W1,
        float* __restrict__ h1, int N_, int K_) {
    __shared__ __align__(32) bhalf xsh[TILE_M][KSTEP];  // 8 KB
    __shared__ __align__(32) bhalf wsh[64][KSTEP];      // 4 KB (W1^T tile)

    const int tid  = threadIdx.x;
    const int lane = tid & 31;
    const int wave = tid >> 5;
    const int rowbase = blockIdx.x * TILE_M;

    // staging roles (constant across K loop)
    const int xr_row = tid >> 1;              // LDS row this thread stages
    const int xr_cb  = (tid & 1) * 16;        // column half (0 or 16)
    int xg_row = rowbase + xr_row;            // clamp: garbage rows are never
    if (xg_row >= N_) xg_row = N_ - 1;        // stored, so any valid row is ok
    const float* xrow = x + (size_t)xg_row * K_;

    const int wn  = tid & 63;                 // output column staged
    const int wkc = (tid >> 6) * 8;           // k-chunk within tile

    v8f zero = {};
    v8f acc[4];
    #pragma unroll
    for (int t = 0; t < 4; ++t) acc[t] = zero;

    const int nk = (K_ + KSTEP - 1) / KSTEP;  // 45 steps (44 full + tail)
    for (int kk = 0; kk < nk; ++kk) {
        const int k0 = kk * KSTEP;
        const bool full = (k0 + KSTEP) <= K_;

        if (full) {
            // --- fast path: unconditional wide loads, no predication ---
            const float* xp = xrow + k0 + xr_cb;
            f4u q0 = *(const f4u*)(xp + 0);
            f4u q1 = *(const f4u*)(xp + 4);
            f4u q2 = *(const f4u*)(xp + 8);
            f4u q3 = *(const f4u*)(xp + 12);
            bhalf xt[16];
            #pragma unroll
            for (int j = 0; j < 4; ++j) {
                xt[j]      = (bhalf)q0[j];
                xt[4 + j]  = (bhalf)q1[j];
                xt[8 + j]  = (bhalf)q2[j];
                xt[12 + j] = (bhalf)q3[j];
            }
            *(v8bf*)&xsh[xr_row][xr_cb]     = *(v8bf*)&xt[0];
            *(v8bf*)&xsh[xr_row][xr_cb + 8] = *(v8bf*)&xt[8];

            const float* wp = W1 + (size_t)(k0 + wkc) * 64 + wn;
            bhalf wt[8];
            #pragma unroll
            for (int j = 0; j < 8; ++j)
                wt[j] = (bhalf)wp[(size_t)j * 64];
            *(v8bf*)&wsh[wn][wkc] = *(v8bf*)&wt[0];
        } else {
            // --- tail path (runs once): per-element k guard ---
            #pragma unroll
            for (int j = 0; j < 16; ++j) {
                const int k = k0 + xr_cb + j;
                xsh[xr_row][xr_cb + j] = (bhalf)((k < K_) ? xrow[k] : 0.0f);
            }
            #pragma unroll
            for (int j = 0; j < 8; ++j) {
                const int k = k0 + wkc + j;
                wsh[wn][wkc + j] = (bhalf)((k < K_) ? W1[(size_t)k * 64 + wn] : 0.0f);
            }
        }
        __syncthreads();

        // --- A fragment: 16x32 bf16 per ISA layout
        //     lanes 0-15: M=lane, K = 0..7 (v0-3) and 16..23 (v4-7)
        //     lanes16-31: M=lane-16, K = 8..15 and 24..31
        const int m  = (wave << 4) + (lane & 15);
        const int hh = (lane >> 4) * 8;
        union { v16bf v; v8bf h[2]; } afrag;
        afrag.h[0] = *(const v8bf*)&xsh[m][hh];
        afrag.h[1] = *(const v8bf*)&xsh[m][16 + hh];

        // --- 4 N-tiles of B (32x16 bf16): lane holds col N=lane&15,
        //     K = (lane>>4)*8 .. +7 and 16+(lane>>4)*8 .. +7 (matches A).
        const int kb = (lane >> 4) * 8;
        #pragma unroll
        for (int t = 0; t < 4; ++t) {
            const int n = (t << 4) + (lane & 15);
            union { v16bf v; v8bf h[2]; } bfrag;
            bfrag.h[0] = *(const v8bf*)&wsh[n][kb];
            bfrag.h[1] = *(const v8bf*)&wsh[n][16 + kb];
            acc[t] = __builtin_amdgcn_wmma_f32_16x16x32_bf16(
                false, afrag.v, false, bfrag.v, (short)0, acc[t], false, false);
        }
        __syncthreads();
    }

    // --- write back: C layout: vgpr r -> M = r + (lane>>4)*8, N = lane&15
    #pragma unroll
    for (int t = 0; t < 4; ++t) {
        #pragma unroll
        for (int r = 0; r < 8; ++r) {
            const int mloc = r + (lane >> 4) * 8;
            const int grow = rowbase + (wave << 4) + mloc;
            const int col  = (t << 4) + (lane & 15);
            if (grow < N_) h1[(size_t)grow * 64 + col] = acc[t][r];
        }
    }
}

// ---------------------------------------------------------------------------
// Attention coefficients layer 1: a_s[n][h] = sum_c h1[n][h][c]*att_src[h][c]
// one thread per (node, head)
// ---------------------------------------------------------------------------
__global__ void attn1_coeff(const float* __restrict__ h1,
                            const float* __restrict__ att_src,
                            const float* __restrict__ att_dst,
                            float* __restrict__ a_s, float* __restrict__ a_d,
                            int N_) {
    int t = blockIdx.x * blockDim.x + threadIdx.x;
    if (t >= N_ * H1) return;
    int n = t >> 3, hd = t & 7;
    const float* hp = h1 + (size_t)n * 64 + hd * HID;
    float s = 0.f, d = 0.f;
    #pragma unroll
    for (int c = 0; c < HID; ++c) {
        s = fmaf(hp[c], att_src[hd * HID + c], s);
        d = fmaf(hp[c], att_dst[hd * HID + c], d);
    }
    a_s[t] = s;
    a_d[t] = d;
}

// ---------------------------------------------------------------------------
// Edge pass 1a: per-dst running max of leaky(e).  One thread per (edge,head).
// Self loops appended: edge id >= E -> src = dst = id - E.
// ---------------------------------------------------------------------------
__global__ void edge1_max(const int* __restrict__ src, const int* __restrict__ dst,
                          const float* __restrict__ a_s, const float* __restrict__ a_d,
                          float* __restrict__ mx, int E_, int N_) {
    long long t = (long long)blockIdx.x * blockDim.x + threadIdx.x;
    long long tot = (long long)(E_ + N_) * H1;
    if (t >= tot) return;
    int e  = (int)(t >> 3), hd = (int)(t & 7);
    int s_, d_;
    if (e < E_) { s_ = src[e]; d_ = dst[e]; } else { s_ = d_ = e - E_; }
    float v = leaky(a_s[s_ * H1 + hd] + a_d[d_ * H1 + hd]);
    atomicMaxF32(&mx[d_ * H1 + hd], v);
}

// ---------------------------------------------------------------------------
// Edge pass 1b: softmax weights + weighted message scatter-add.
// ---------------------------------------------------------------------------
__global__ void edge1_scatter(const int* __restrict__ src, const int* __restrict__ dst,
                              const float* __restrict__ a_s, const float* __restrict__ a_d,
                              const float* __restrict__ mx, const float* __restrict__ h1,
                              float* __restrict__ denom, float* __restrict__ acc,
                              int E_, int N_) {
    long long t = (long long)blockIdx.x * blockDim.x + threadIdx.x;
    long long tot = (long long)(E_ + N_) * H1;
    if (t >= tot) return;
    int e  = (int)(t >> 3), hd = (int)(t & 7);
    int s_, d_;
    if (e < E_) { s_ = src[e]; d_ = dst[e]; } else { s_ = d_ = e - E_; }
    float v = leaky(a_s[s_ * H1 + hd] + a_d[d_ * H1 + hd]);
    float w = __expf(v - mx[d_ * H1 + hd]);
    atomicAdd(&denom[d_ * H1 + hd], w);
    const float4* hp = (const float4*)(h1 + (size_t)s_ * 64 + hd * HID);
    float4 p0 = hp[0], p1 = hp[1];
    float* ap = acc + (size_t)d_ * 64 + hd * HID;
    atomicAdd(ap + 0, p0.x * w); atomicAdd(ap + 1, p0.y * w);
    atomicAdd(ap + 2, p0.z * w); atomicAdd(ap + 3, p0.w * w);
    atomicAdd(ap + 4, p1.x * w); atomicAdd(ap + 5, p1.y * w);
    atomicAdd(ap + 6, p1.z * w); atomicAdd(ap + 7, p1.w * w);
}

// ---------------------------------------------------------------------------
// Finalize layer 1 (normalize, head-mean, +bias, ELU) fused with the tiny
// layer-2 projection (8x7) and its attention coefficients. One thread/node.
// ---------------------------------------------------------------------------
__global__ void finalize1_proj2(const float* __restrict__ acc1,
                                const float* __restrict__ denom1,
                                const float* __restrict__ b1,
                                const float* __restrict__ W2,
                                const float* __restrict__ att_src2,
                                const float* __restrict__ att_dst2,
                                float* __restrict__ h2,
                                float* __restrict__ as2, float* __restrict__ ad2,
                                int N_) {
    int n = blockIdx.x * blockDim.x + threadIdx.x;
    if (n >= N_) return;
    float inv_den[H1];
    #pragma unroll
    for (int h = 0; h < H1; ++h)
        inv_den[h] = 1.0f / (denom1[n * H1 + h] + 1e-16f);
    float hin[HID];
    #pragma unroll
    for (int c = 0; c < HID; ++c) {
        float s = 0.f;
        #pragma unroll
        for (int h = 0; h < H1; ++h)
            s = fmaf(acc1[(size_t)n * 64 + h * HID + c], inv_den[h], s);
        s = s * (1.0f / H1) + b1[c];
        hin[c] = s > 0.f ? s : (__expf(s) - 1.0f);   // ELU
    }
    float as = 0.f, ad = 0.f;
    #pragma unroll
    for (int c2 = 0; c2 < NCLS; ++c2) {
        float o = 0.f;
        #pragma unroll
        for (int c = 0; c < HID; ++c)
            o = fmaf(hin[c], W2[c * NCLS + c2], o);
        h2[(size_t)n * NCLS + c2] = o;
        as = fmaf(o, att_src2[c2], as);
        ad = fmaf(o, att_dst2[c2], ad);
    }
    as2[n] = as;
    ad2[n] = ad;
}

// ---------------------------------------------------------------------------
// Edge passes, layer 2 (single head). One thread per edge.
// ---------------------------------------------------------------------------
__global__ void edge2_max(const int* __restrict__ src, const int* __restrict__ dst,
                          const float* __restrict__ as2, const float* __restrict__ ad2,
                          float* __restrict__ mx2, int E_, int N_) {
    int e = blockIdx.x * blockDim.x + threadIdx.x;
    if (e >= E_ + N_) return;
    int s_, d_;
    if (e < E_) { s_ = src[e]; d_ = dst[e]; } else { s_ = d_ = e - E_; }
    atomicMaxF32(&mx2[d_], leaky(as2[s_] + ad2[d_]));
}

__global__ void edge2_scatter(const int* __restrict__ src, const int* __restrict__ dst,
                              const float* __restrict__ as2, const float* __restrict__ ad2,
                              const float* __restrict__ mx2, const float* __restrict__ h2,
                              float* __restrict__ den2, float* __restrict__ acc2,
                              int E_, int N_) {
    int e = blockIdx.x * blockDim.x + threadIdx.x;
    if (e >= E_ + N_) return;
    int s_, d_;
    if (e < E_) { s_ = src[e]; d_ = dst[e]; } else { s_ = d_ = e - E_; }
    float w = __expf(leaky(as2[s_] + ad2[d_]) - mx2[d_]);
    atomicAdd(&den2[d_], w);
    const float* hp = h2 + (size_t)s_ * NCLS;
    float* ap = acc2 + (size_t)d_ * NCLS;
    #pragma unroll
    for (int c = 0; c < NCLS; ++c)
        atomicAdd(ap + c, hp[c] * w);
}

// ---------------------------------------------------------------------------
// Finalize: normalize, +b2, log_softmax. One thread per node.
// ---------------------------------------------------------------------------
__global__ void finalize2(const float* __restrict__ acc2, const float* __restrict__ den2,
                          const float* __restrict__ b2, float* __restrict__ out,
                          int N_) {
    int n = blockIdx.x * blockDim.x + threadIdx.x;
    if (n >= N_) return;
    float inv = 1.0f / (den2[n] + 1e-16f);
    float o[NCLS];
    float m = -INFINITY;
    #pragma unroll
    for (int c = 0; c < NCLS; ++c) {
        o[c] = acc2[(size_t)n * NCLS + c] * inv + b2[c];
        m = fmaxf(m, o[c]);
    }
    float s = 0.f;
    #pragma unroll
    for (int c = 0; c < NCLS; ++c) s += __expf(o[c] - m);
    float lse = m + logf(s);
    #pragma unroll
    for (int c = 0; c < NCLS; ++c)
        out[(size_t)n * NCLS + c] = o[c] - lse;
}

// ---------------------------------------------------------------------------
// Launcher
// ---------------------------------------------------------------------------
extern "C" void kernel_launch(void* const* d_in, const int* in_sizes, int n_in,
                              void* d_out, int out_size, void* d_ws, size_t ws_size,
                              hipStream_t stream) {
    const float* x        = (const float*)d_in[0];
    const int*   eidx     = (const int*)  d_in[1];
    const float* W1       = (const float*)d_in[2];
    const float* att_src1 = (const float*)d_in[3];
    const float* att_dst1 = (const float*)d_in[4];
    const float* b1       = (const float*)d_in[5];
    const float* W2       = (const float*)d_in[6];
    const float* att_src2 = (const float*)d_in[7];
    const float* att_dst2 = (const float*)d_in[8];
    const float* b2       = (const float*)d_in[9];
    float* out = (float*)d_out;

    const int N_ = in_sizes[0] / F_IN;
    const int E_ = in_sizes[1] / 2;
    const int*  src = eidx;
    const int*  dst = eidx + E_;

    // workspace carve-up (fp32)
    float* w = (float*)d_ws;
    float* h1    = w;                 w += (size_t)N_ * 64;
    float* a_s1  = w;                 w += (size_t)N_ * H1;
    float* a_d1  = w;                 w += (size_t)N_ * H1;
    float* mx1   = w;                 w += (size_t)N_ * H1;
    float* den1  = w;                 w += (size_t)N_ * H1;
    float* acc1  = w;                 w += (size_t)N_ * 64;
    float* h2    = w;                 w += (size_t)N_ * NCLS;
    float* as2   = w;                 w += (size_t)N_;
    float* ad2   = w;                 w += (size_t)N_;
    float* mx2   = w;                 w += (size_t)N_;
    float* den2  = w;                 w += (size_t)N_;
    float* acc2  = w;                 w += (size_t)N_ * NCLS;

    const int BT = 256;
    auto blocks = [&](long long n) { return (int)((n + BT - 1) / BT); };

    // init accumulators / maxes
    fill_kernel<<<blocks((long long)N_ * H1), BT, 0, stream>>>(mx1,  (long long)N_ * H1, -INFINITY);
    fill_kernel<<<blocks((long long)N_ * H1), BT, 0, stream>>>(den1, (long long)N_ * H1, 0.0f);
    fill_kernel<<<blocks((long long)N_ * 64), BT, 0, stream>>>(acc1, (long long)N_ * 64, 0.0f);
    fill_kernel<<<blocks(N_), BT, 0, stream>>>(mx2,  N_, -INFINITY);
    fill_kernel<<<blocks(N_), BT, 0, stream>>>(den2, N_, 0.0f);
    fill_kernel<<<blocks((long long)N_ * NCLS), BT, 0, stream>>>(acc2, (long long)N_ * NCLS, 0.0f);

    // layer 1 projection via bf16 WMMA
    gemm1_wmma<<<(N_ + TILE_M - 1) / TILE_M, 256, 0, stream>>>(x, W1, h1, N_, F_IN);

    // attention coefficients
    attn1_coeff<<<blocks((long long)N_ * H1), BT, 0, stream>>>(h1, att_src1, att_dst1, a_s1, a_d1, N_);

    // segment softmax + scatter, layer 1
    long long et1 = (long long)(E_ + N_) * H1;
    edge1_max    <<<blocks(et1), BT, 0, stream>>>(src, dst, a_s1, a_d1, mx1, E_, N_);
    edge1_scatter<<<blocks(et1), BT, 0, stream>>>(src, dst, a_s1, a_d1, mx1, h1, den1, acc1, E_, N_);

    // finalize layer 1 + layer-2 projection + coefficients
    finalize1_proj2<<<blocks(N_), BT, 0, stream>>>(acc1, den1, b1, W2, att_src2, att_dst2,
                                                   h2, as2, ad2, N_);

    // segment softmax + scatter, layer 2
    edge2_max    <<<blocks(E_ + N_), BT, 0, stream>>>(src, dst, as2, ad2, mx2, E_, N_);
    edge2_scatter<<<blocks(E_ + N_), BT, 0, stream>>>(src, dst, as2, ad2, mx2, h2, den2, acc2, E_, N_);

    // log_softmax output
    finalize2<<<blocks(N_), BT, 0, stream>>>(acc2, den2, b2, out, N_);
}
